// InterModalAttention_1176821039548
// MI455X (gfx1250) — matrix operand order repaired and checked by reading, not compiled, not run on hardware
//
#include <hip/hip_runtime.h>
#include <stdint.h>
#include <stddef.h>

// ---------- CDNA5 WMMA types ----------
typedef __attribute__((ext_vector_type(16))) __bf16 v16bf;
typedef __attribute__((ext_vector_type(8)))  float  v8f;
typedef __attribute__((ext_vector_type(4)))  int    v4i;

// gfx1250 async global->LDS copy path (probe via __has_builtin; fallback = reg staging)
#if defined(__AMDGCN__) && __has_builtin(__builtin_amdgcn_global_load_async_to_lds_b128)
#define USE_ASYNC_LDS 1
typedef __attribute__((address_space(1))) v4i* as1_v4i;   // global int4*
typedef __attribute__((address_space(3))) v4i* as3_v4i;   // LDS    int4*
#else
#define USE_ASYNC_LDS 0
#endif

__device__ __forceinline__ void wait_async0() {
#if USE_ASYNC_LDS
#if __has_builtin(__builtin_amdgcn_s_wait_asynccnt)
  __builtin_amdgcn_s_wait_asynccnt(0);
#else
  asm volatile("s_wait_asynccnt 0x0" ::: "memory");
#endif
#endif
}

// float -> bf16 (round-to-nearest-even), bit math only
__device__ __forceinline__ unsigned short f2bf(float f) {
  union { float f; unsigned int u; } x; x.f = f;
  unsigned int r = x.u + 0x7fffu + ((x.u >> 16) & 1u);
  return (unsigned short)(r >> 16);
}

// ---------- 1) f32 -> bf16 convert (float4 -> ushort4) ----------
__global__ void k_cvt_bf16(const float* __restrict__ in, unsigned short* __restrict__ out, int n4) {
  int i = blockIdx.x * blockDim.x + threadIdx.x;
  const int stride = gridDim.x * blockDim.x;
  for (; i < n4; i += stride) {
    float4 f = ((const float4*)in)[i];
    ushort4 o;
    o.x = f2bf(f.x); o.y = f2bf(f.y); o.z = f2bf(f.z); o.w = f2bf(f.w);
    ((ushort4*)out)[i] = o;
  }
}

// ---------- 2) TN GEMM: C[M,N] = A[M,K] * B[N,K]^T (+bias) ----------
// MODE 0: bf16 out + bias (Q/K projections)
// MODE 1: f32 out, no bias (scores, P*V)
// MODE 2: bf16 out + bias, transposed per batch: CbT[b][N][rowsPB] (V projection)
// Block tile 128x128, BK=32, 8 waves (4x2), wave tile 32x64 = 2x4 WMMA frags.
// Ping-pong LDS with compile-time buffer ids (K % 64 == 0), async-LDS staging when available.
template<int MODE>
__global__ __launch_bounds__(256) void k_gemm_tn(
    const unsigned short* __restrict__ A,
    const unsigned short* __restrict__ Bm,
    const float* __restrict__ bias,
    unsigned short* __restrict__ Cb,
    float* __restrict__ Cf,
    int M, int N, int K, int rowsPB)
{
  constexpr int BM = 128, BN = 128, BK = 32, LDS = 40;  // 80B rows: 16B aligned, conflict-free
  __shared__ unsigned short sA[2][BM][LDS];
  __shared__ unsigned short sB[2][BN][LDS];

  const int tid  = threadIdx.x;
  const int lane = tid & 31;
  const int wave = tid >> 5;
  const int wm   = wave & 3;       // 0..3 -> 32-row strip
  const int wn   = wave >> 2;      // 0..1 -> 64-col strip
  const int m0   = blockIdx.x * BM;
  const int n0   = blockIdx.y * BN;

  v8f vzero = {};
  v8f acc[2][4];
#pragma unroll
  for (int i = 0; i < 2; ++i)
#pragma unroll
    for (int j = 0; j < 4; ++j) acc[i][j] = vzero;

  // staging: 128 rows x 32 cols per matrix, 16 bf16 (2 x b128) per thread
  const int s_row = tid >> 1;            // 0..127
  const int s_col = (tid & 1) * 16;      // 0 / 16
  const unsigned short* gA = A  + (size_t)(m0 + s_row) * K + s_col;
  const unsigned short* gB = Bm + (size_t)(n0 + s_row) * K + s_col;

  const int mr = lane & 15;              // row/col within 16x16 frag
  const int kh = lane >> 4;              // K-half selector

  uint4 ra0, ra1, rb0, rb1;              // fallback staging registers

  auto stage_pre = [&](int kk, int bb) {
#if USE_ASYNC_LDS
    __builtin_amdgcn_global_load_async_to_lds_b128(
        (as1_v4i)(gA + kk),     (as3_v4i)&sA[bb][s_row][s_col],     0, 0);
    __builtin_amdgcn_global_load_async_to_lds_b128(
        (as1_v4i)(gA + kk + 8), (as3_v4i)&sA[bb][s_row][s_col + 8], 0, 0);
    __builtin_amdgcn_global_load_async_to_lds_b128(
        (as1_v4i)(gB + kk),     (as3_v4i)&sB[bb][s_row][s_col],     0, 0);
    __builtin_amdgcn_global_load_async_to_lds_b128(
        (as1_v4i)(gB + kk + 8), (as3_v4i)&sB[bb][s_row][s_col + 8], 0, 0);
#else
    const uint4* a = (const uint4*)(gA + kk);
    const uint4* b = (const uint4*)(gB + kk);
    ra0 = a[0]; ra1 = a[1]; rb0 = b[0]; rb1 = b[1];
#endif
    if (kk + BK < K) {
      __builtin_prefetch(gA + kk + BK, 0, 1);
      __builtin_prefetch(gB + kk + BK, 0, 1);
    }
  };

  auto stage_post = [&](int bb) {
#if USE_ASYNC_LDS
    (void)bb;
    wait_async0();
#else
    *(uint4*)&sA[bb][s_row][s_col]     = ra0;
    *(uint4*)&sA[bb][s_row][s_col + 8] = ra1;
    *(uint4*)&sB[bb][s_row][s_col]     = rb0;
    *(uint4*)&sB[bb][s_row][s_col + 8] = rb1;
#endif
  };

  auto compute = [&](int bb) {
    union { v16bf v; uint4 q[2]; } af[2], bfr[4];
#pragma unroll
    for (int i = 0; i < 2; ++i) {
      const unsigned short* p = &sA[bb][wm * 32 + i * 16 + mr][kh * 8];
      af[i].q[0] = *(const uint4*)p;          // K = kh*8 + 0..7
      af[i].q[1] = *(const uint4*)(p + 16);   // K = 16 + kh*8 + 0..7
    }
#pragma unroll
    for (int j = 0; j < 4; ++j) {
      const unsigned short* p = &sB[bb][wn * 64 + j * 16 + mr][kh * 8];
      bfr[j].q[0] = *(const uint4*)p;
      bfr[j].q[1] = *(const uint4*)(p + 16);
    }
#pragma unroll
    for (int i = 0; i < 2; ++i)
#pragma unroll
      for (int j = 0; j < 4; ++j)
        acc[i][j] = __builtin_amdgcn_wmma_f32_16x16x32_bf16(
            false, af[i].v, false, bfr[j].v, (short)0, acc[i][j], false, false);
  };

  // prologue: tile 0 -> buffer 0
  stage_pre(0, 0);
  stage_post(0);
  __syncthreads();

  for (int k0 = 0; k0 < K; k0 += 2 * BK) {   // K % 64 == 0
    // half 1: compute buf0 (k0), stage k0+BK -> buf1
    stage_pre(k0 + BK, 1);
    compute(0);
    stage_post(1);
    __syncthreads();
    // half 2: compute buf1 (k0+BK), stage k0+2BK -> buf0
    const bool more = (k0 + 2 * BK) < K;
    if (more) stage_pre(k0 + 2 * BK, 0);
    compute(1);
    if (more) stage_post(0);
    __syncthreads();
  }

  // C frag layout: VGPR r of lane l = C[M = r + 8*(l>>4)][N = l&15]
  const int ccol = lane & 15;
  const int crh  = lane >> 4;
#pragma unroll
  for (int i = 0; i < 2; ++i)
#pragma unroll
    for (int j = 0; j < 4; ++j) {
      const int gn = n0 + wn * 64 + j * 16 + ccol;
      const float bv = (MODE != 1) ? bias[gn] : 0.0f;
      const int gmBase = m0 + wm * 32 + i * 16 + crh * 8;
      if (MODE == 2) {
        // transposed store: 8 consecutive rows -> contiguous bf16, one b128 store
        const int b    = gmBase / rowsPB;
        const int mrow = gmBase % rowsPB;
        ushort4 pk[2];
        unsigned short* t = (unsigned short*)pk;
#pragma unroll
        for (int r = 0; r < 8; ++r) t[r] = f2bf(acc[i][j][r] + bv);
        *(uint4*)&Cb[(size_t)b * N * rowsPB + (size_t)gn * rowsPB + mrow] = *(uint4*)pk;
      } else {
#pragma unroll
        for (int r = 0; r < 8; ++r) {
          const int gm = gmBase + r;
          const float v = acc[i][j][r] + bv;
          if (MODE == 0) Cb[(size_t)gm * N + gn] = f2bf(v);
          else           Cf[(size_t)gm * N + gn] = v;
        }
      }
    }
}

// ---------- 3) row softmax: P_bf16 = softmax(scale * S_f32) ----------
__global__ __launch_bounds__(256) void k_softmax_rows(
    const float* __restrict__ S, unsigned short* __restrict__ P, int ncol, float scale)
{
  const int row = blockIdx.x;
  const float* x = S + (size_t)row * ncol;
  unsigned short* p = P + (size_t)row * ncol;
  const int tid = threadIdx.x;
  const int nper = ncol / 256;   // 8 for ncol=2048

  float v[8];
  float mx = -3.4e38f;
#pragma unroll
  for (int i = 0; i < 8; ++i)
    if (i < nper) { v[i] = x[tid + i * 256]; mx = fmaxf(mx, v[i]); }
#pragma unroll
  for (int off = 16; off > 0; off >>= 1) mx = fmaxf(mx, __shfl_xor(mx, off, 32));

  __shared__ float red[8];
  if ((tid & 31) == 0) red[tid >> 5] = mx;
  __syncthreads();
  float rmax = red[0];
#pragma unroll
  for (int w = 1; w < 8; ++w) rmax = fmaxf(rmax, red[w]);
  __syncthreads();

  float s = 0.0f;
#pragma unroll
  for (int i = 0; i < 8; ++i)
    if (i < nper) { v[i] = __expf((v[i] - rmax) * scale); s += v[i]; }
#pragma unroll
  for (int off = 16; off > 0; off >>= 1) s += __shfl_xor(s, off, 32);
  if ((tid & 31) == 0) red[tid >> 5] = s;
  __syncthreads();
  float rsum = 0.0f;
#pragma unroll
  for (int w = 0; w < 8; ++w) rsum += red[w];
  const float inv = 1.0f / rsum;

#pragma unroll
  for (int i = 0; i < 8; ++i)
    if (i < nper) p[tid + i * 256] = f2bf(v[i] * inv);
}

// ---------- host launch ----------
extern "C" void kernel_launch(void* const* d_in, const int* in_sizes, int n_in,
                              void* d_out, int out_size, void* d_ws, size_t ws_size,
                              hipStream_t stream) {
  (void)in_sizes; (void)n_in; (void)out_size; (void)ws_size;
  const int B = 4, S = 2048, D = 1024;
  const int MS = B * S;               // 8192

  const float* x  = (const float*)d_in[0];
  const float* Wq = (const float*)d_in[1];
  const float* bq = (const float*)d_in[2];
  const float* Wk = (const float*)d_in[3];
  const float* bk = (const float*)d_in[4];
  const float* Wv = (const float*)d_in[5];
  const float* bv = (const float*)d_in[6];
  float* out = (float*)d_out;

  char* ws = (char*)d_ws;
  const size_t MB = 1u << 20;
  unsigned short* xb  = (unsigned short*)(ws + 0 * MB);    // 16 MB  bf16 x     [8192,1024]
  unsigned short* qb  = (unsigned short*)(ws + 16 * MB);   // 16 MB  bf16 Q     [8192,1024]
  unsigned short* kb  = (unsigned short*)(ws + 32 * MB);   // 16 MB  bf16 K     [8192,1024]
  unsigned short* vbT = (unsigned short*)(ws + 48 * MB);   // 16 MB  bf16 V^T   [4][1024][2048]
  unsigned short* wqb = (unsigned short*)(ws + 64 * MB);   //  2 MB
  unsigned short* wkb = (unsigned short*)(ws + 66 * MB);   //  2 MB
  unsigned short* wvb = (unsigned short*)(ws + 68 * MB);   //  2 MB
  float*          sc  = (float*)(ws + 70 * MB);            // 16 MB  f32 scores [2048,2048]
  unsigned short* pb  = (unsigned short*)(ws + 86 * MB);   //  8 MB  bf16 P     [2048,2048]

  // 1) convert inputs to bf16
  k_cvt_bf16<<<2048, 256, 0, stream>>>(x,  xb,  MS * D / 4);
  k_cvt_bf16<<<512,  256, 0, stream>>>(Wq, wqb, D * D / 4);
  k_cvt_bf16<<<512,  256, 0, stream>>>(Wk, wkb, D * D / 4);
  k_cvt_bf16<<<512,  256, 0, stream>>>(Wv, wvb, D * D / 4);

  // 2) QKV projections: [8192,1024] x [1024,1024]^T + bias
  dim3 gqkv(MS / 128, D / 128);
  k_gemm_tn<0><<<gqkv, 256, 0, stream>>>(xb, wqb, bq, qb,  nullptr, MS, D, D, S);
  k_gemm_tn<0><<<gqkv, 256, 0, stream>>>(xb, wkb, bk, kb,  nullptr, MS, D, D, S);
  k_gemm_tn<2><<<gqkv, 256, 0, stream>>>(xb, wvb, bv, vbT, nullptr, MS, D, D, S); // V -> [b][D][S]

  // 3) per-batch attention
  const float scale = 0.03125f;   // 1/sqrt(1024)
  for (int b = 0; b < B; ++b) {
    const size_t off = (size_t)b * S * D;
    dim3 gsc(S / 128, S / 128);
    k_gemm_tn<1><<<gsc, 256, 0, stream>>>(qb + off, kb + off, nullptr,
                                          nullptr, sc, S, S, D, S);
    k_softmax_rows<<<S, 256, 0, stream>>>(sc, pb, S, scale);
    // P[2048,2048] x (V^T[b])[1024,2048]^T -> out[2048,1024]
    dim3 gpv(S / 128, D / 128);
    k_gemm_tn<1><<<gpv, 256, 0, stream>>>(pb, vbT + off, nullptr,
                                          nullptr, out + off, S, D, S, S);
  }
}
